// BiCoRLayer_47648367181894
// MI455X (gfx1250) — matrix operand order repaired and checked by reading, not compiled, not run on hardware
//
#include <hip/hip_runtime.h>
#include <hip/hip_bf16.h>
#include <math.h>

#define B_     2
#define NCLS   150
#define CF_    256
#define CD_    256
#define S_     16384      // H*W
#define KP     160        // class dim padded to multiple of 32 for WMMA K
#define KSEL   4096       // top-k count: max(64, 0.25*16384)
#define KSPLIT 8          // split-K chunks for the Cn GEMM
#define EPS_   1e-6f
#define LNEPS  1e-5f
#define LAM_   0.5f

typedef _Float16 h16;
typedef __attribute__((ext_vector_type(16))) _Float16 v16h;
typedef __attribute__((ext_vector_type(8)))  _Float16 v8h;
typedef __attribute__((ext_vector_type(8)))  float    v8f;

__device__ __forceinline__ v16h load_afrag(const h16* __restrict__ p) {
    v8h lo = *(const v8h*)p;
    v8h hi = *(const v8h*)(p + 8);
    return __builtin_shufflevector(lo, hi, 0,1,2,3,4,5,6,7,8,9,10,11,12,13,14,15);
}

__device__ __forceinline__ void wmma_store16(v8f acc, float* __restrict__ Cf,
                                             h16* __restrict__ Ch, size_t cbase,
                                             int m0t, int ncol, int M, int ldc,
                                             float alpha, int mode, int lane)
{
    int mb = m0t + ((lane & 16) ? 8 : 0);
    #pragma unroll
    for (int r = 0; r < 8; ++r) {
        int m = mb + r;
        if (m < M) {
            float v = acc[r] * alpha;
            size_t off = cbase + (size_t)m * ldc + ncol;
            if (mode & 1) Cf[off] = v;
            if (mode & 2) Ch[off] = (h16)v;
        }
    }
}

// ---------------------------------------------------------------------------
// Simple WMMA GEMM (1 tile/wave): C[MxN] = alpha * A[MxK] * Bt[NxK]^T
// Ragged M handled by clamping the A row (no exec divergence in the k-loop);
// out-of-range D rows are masked at store. K % 32 == 0.
// mode bit0: f32 store; bit1: f16 store.
// ---------------------------------------------------------------------------
__global__ void __launch_bounds__(256)
wmma_gemm_nt(const h16* __restrict__ A, const h16* __restrict__ Bt,
             float* __restrict__ Cf, h16* __restrict__ Ch,
             int M, int N, int K, int lda, int ldb, int ldc,
             long strideA, long strideB, long strideC, float alpha, int mode)
{
    const int bz = blockIdx.z;
    const h16* Ab = A + (size_t)bz * strideA;
    const h16* Bb = Bt + (size_t)bz * strideB;
    const int lane = threadIdx.x & 31;
    const int wave = threadIdx.x >> 5;
    const int n0 = (blockIdx.x * 8 + wave) * 16;
    const int m0 = blockIdx.y * 16;
    if (n0 >= N || m0 >= M) return;               // wave-uniform

    int mrow = m0 + (lane & 15);
    mrow = (mrow < M) ? mrow : (M - 1);           // clamp: no divergence in loop
    const int kbA = (lane & 16) ? 16 : 0;
    const int nrow = n0 + (lane & 15);
    const int kbB = (lane & 16) ? 16 : 0;
    const h16* ap = Ab + (size_t)mrow * lda + kbA;
    const h16* bp = Bb + (size_t)nrow * ldb + kbB;

    v8f acc = {};
    for (int k0 = 0; k0 < K; k0 += 32) {
        v16h afr = load_afrag(ap + k0);
        v16h bfr = *(const v16h*)(bp + k0);
        acc = __builtin_amdgcn_wmma_f32_16x16x32_f16(
            false, afr, false, bfr, (short)0, acc, false, false);
    }
    wmma_store16(acc, Cf, Ch, (size_t)bz * strideC, m0, nrow, M, ldc, alpha, mode, lane);
}

// ---------------------------------------------------------------------------
// 2x2 register-blocked WMMA GEMM: each wave computes a 32x32 C macro-tile
// (4 WMMAs per k-step from 2 A-frags + 2 B-frags -> 2x arithmetic intensity).
// Block = 8 waves arranged 2(M) x 4(N): covers 64 rows x 128 cols.
// Optional deterministic split-K: blockIdx.z = batch*ksplit + chunk, partial
// results land at chunk*strideKc (mode should be 1 when ksplit > 1).
// ---------------------------------------------------------------------------
__global__ void __launch_bounds__(256)
wmma_gemm_nt_2x2(const h16* __restrict__ A, const h16* __restrict__ Bt,
                 float* __restrict__ Cf, h16* __restrict__ Ch,
                 int M, int N, int K, int lda, int ldb, int ldc,
                 long strideA, long strideB, long strideC, long strideKc,
                 float alpha, int mode, int ksplit)
{
    const int bz    = blockIdx.z;
    const int chunk = bz % ksplit;
    const int batch = bz / ksplit;
    const int kLen  = K / ksplit;
    const int kLo   = chunk * kLen;

    const h16* Ab = A + (size_t)batch * strideA;
    const h16* Bb = Bt + (size_t)batch * strideB;
    const int lane = threadIdx.x & 31;
    const int wave = threadIdx.x >> 5;
    const int m0 = blockIdx.y * 64 + (wave >> 2) * 32;   // 2 M-tiles: m0, m0+16
    const int n0 = blockIdx.x * 128 + (wave & 3) * 32;   // 2 N-tiles: n0, n0+16
    if (m0 >= M) return;                                  // wave-uniform

    const int kbA = (lane & 16) ? 16 : 0;
    const int kbB = (lane & 16) ? 16 : 0;
    int mr0 = m0 + (lane & 15);        mr0 = (mr0 < M) ? mr0 : (M - 1);
    int mr1 = m0 + 16 + (lane & 15);   mr1 = (mr1 < M) ? mr1 : (M - 1);
    const h16* ap0 = Ab + (size_t)mr0 * lda + kbA;
    const h16* ap1 = Ab + (size_t)mr1 * lda + kbA;
    const h16* bp0 = Bb + (size_t)(n0 + (lane & 15)) * ldb + kbB;
    const h16* bp1 = bp0 + (size_t)16 * ldb;

    v8f acc00 = {}, acc01 = {}, acc10 = {}, acc11 = {};
    for (int k0 = kLo; k0 < kLo + kLen; k0 += 32) {
        v16h a0 = load_afrag(ap0 + k0);
        v16h a1 = load_afrag(ap1 + k0);
        v16h b0 = *(const v16h*)(bp0 + k0);
        v16h b1 = *(const v16h*)(bp1 + k0);
        acc00 = __builtin_amdgcn_wmma_f32_16x16x32_f16(false, a0, false, b0, (short)0, acc00, false, false);
        acc01 = __builtin_amdgcn_wmma_f32_16x16x32_f16(false, a0, false, b1, (short)0, acc01, false, false);
        acc10 = __builtin_amdgcn_wmma_f32_16x16x32_f16(false, a1, false, b0, (short)0, acc10, false, false);
        acc11 = __builtin_amdgcn_wmma_f32_16x16x32_f16(false, a1, false, b1, (short)0, acc11, false, false);
    }

    const size_t cbase = (size_t)batch * strideC + (size_t)chunk * strideKc;
    const int nc0 = n0 + (lane & 15);
    wmma_store16(acc00, Cf, Ch, cbase, m0,      nc0,      M, ldc, alpha, mode, lane);
    wmma_store16(acc01, Cf, Ch, cbase, m0,      nc0 + 16, M, ldc, alpha, mode, lane);
    wmma_store16(acc10, Cf, Ch, cbase, m0 + 16, nc0,      M, ldc, alpha, mode, lane);
    wmma_store16(acc11, Cf, Ch, cbase, m0 + 16, nc0 + 16, M, ldc, alpha, mode, lane);
}

// sum split-K partials -> f32 + f16
__global__ void reduce_partials_kernel(const float* __restrict__ part, float* __restrict__ of,
                                       h16* __restrict__ oh, int total, int chunks, long cstride)
{
    int idx = blockIdx.x * blockDim.x + threadIdx.x;
    if (idx >= total) return;
    float s = 0.f;
    for (int c = 0; c < chunks; ++c) s += part[(size_t)c * cstride + idx];
    of[idx] = s;
    oh[idx] = (h16)s;
}

// ---------------------------------------------------------------------------
// Helpers / elementwise kernels
// ---------------------------------------------------------------------------
__device__ __forceinline__ float block_reduce_sum_256(float v, float* red) {
    int t = threadIdx.x;
    red[t] = v;
    __syncthreads();
    #pragma unroll
    for (int off = 128; off > 0; off >>= 1) {
        if (t < off) red[t] += red[t + off];
        __syncthreads();
    }
    float r = red[0];
    __syncthreads();
    return r;
}

__global__ void cvt_f32_to_f16_kernel(const float* __restrict__ x, h16* __restrict__ y, int n) {
    int i = blockIdx.x * blockDim.x + threadIdx.x;
    if (i < n) y[i] = (h16)x[i];
}

__global__ void __launch_bounds__(256)
ln_rows_kernel(const float* __restrict__ x, const float* __restrict__ g,
               const float* __restrict__ b, float* __restrict__ yf,
               h16* __restrict__ yh, int D)
{
    __shared__ float red[256];
    size_t row = blockIdx.x;
    const float* xr = x + row * D;
    float v = xr[threadIdx.x];
    float mean = block_reduce_sum_256(v, red) * (1.f / D);
    float d = v - mean;
    float var = block_reduce_sum_256(d * d, red) * (1.f / D);
    float y = d * rsqrtf(var + LNEPS) * g[threadIdx.x] + b[threadIdx.x];
    yf[row * D + threadIdx.x] = y;
    yh[row * D + threadIdx.x] = (h16)y;
}

// Channel LN of F (B,CF,S) per pixel -> Fn transposed (B,S,CF) f16.
// Single fused stats pass (sum + sumsq), vectorized v8h output stores.
__global__ void ln_chan_kernel(const float* __restrict__ F, const float* __restrict__ g,
                               const float* __restrict__ bb, h16* __restrict__ Fnt, int total)
{
    int idx = blockIdx.x * blockDim.x + threadIdx.x;
    if (idx >= total) return;                   // total = B_*S_
    int b = idx / S_;
    int s = idx - b * S_;
    const float* base = F + ((size_t)b * CF_) * S_ + s;
    float sum = 0.f, sq = 0.f;
    for (int c = 0; c < CF_; ++c) {
        float x = base[(size_t)c * S_];
        sum += x;
        sq  += x * x;
    }
    float mean = sum * (1.f / CF_);
    float var = sq * (1.f / CF_) - mean * mean;
    var = (var > 0.f) ? var : 0.f;
    float inv = rsqrtf(var + LNEPS);
    h16* out = Fnt + (size_t)idx * CF_;
    for (int c0 = 0; c0 < CF_; c0 += 8) {
        v8h v;
        #pragma unroll
        for (int u = 0; u < 8; ++u) {
            int c = c0 + u;
            v[u] = (h16)((base[(size_t)c * S_] - mean) * inv * g[c] + bb[c]);
        }
        *(v8h*)(out + c0) = v;
    }
}

__device__ __forceinline__ unsigned int fkey(float f) {   // order-preserving f32 key
    unsigned int u = __float_as_uint(f);
    return (u & 0x80000000u) ? ~u : (u | 0x80000000u);
}

// Exact top-k (k=4096) masked sigmoid-weight normalization per (b,n) row.
// Keys for all 16384 elements are cached in registers (64/thread), so the
// 32-step bitwise threshold search runs entirely out of VGPRs (no re-reads).
__global__ void __launch_bounds__(256)
maskw_kernel(const float* __restrict__ logits, h16* __restrict__ Wm)
{
    __shared__ float red[256];
    const size_t row = blockIdx.x;              // b*NCLS + n
    const float* src = logits + row * S_;

    unsigned int key[64];
    #pragma unroll
    for (int j = 0; j < 64; ++j)
        key[j] = fkey(src[threadIdx.x + 256 * j]);

    unsigned int thr = 0u;
    for (int bit = 31; bit >= 0; --bit) {
        unsigned int cand = thr | (1u << bit);
        float cnt = 0.f;
        #pragma unroll
        for (int j = 0; j < 64; ++j)
            cnt += (key[j] >= cand) ? 1.f : 0.f;
        float tot = block_reduce_sum_256(cnt, red);
        if (tot >= (float)KSEL) thr = cand;     // uniform decision
    }

    float swl = 0.f, sal = 0.f;
    #pragma unroll
    for (int j = 0; j < 64; ++j) {
        float sc = src[threadIdx.x + 256 * j];
        float hv = 1.f / (1.f + expf(-sc));
        sal += hv;
        if (key[j] >= thr) swl += hv;
    }
    float sw = block_reduce_sum_256(swl, red);
    float sa = block_reduce_sum_256(sal, red);
    bool fb = (sw <= EPS_);
    float inv = fb ? 1.f / (sa + EPS_) : 1.f / (sw + EPS_);
    h16* dst = Wm + row * S_;
    #pragma unroll
    for (int j = 0; j < 64; ++j) {
        int i = threadIdx.x + 256 * j;
        float hv = 1.f / (1.f + expf(-src[i]));
        float wv = fb ? hv * inv : ((key[j] >= thr) ? hv * inv : 0.f);
        dst[i] = (h16)wv;
    }
}

// Online softmax over classes per pixel; output transposed+padded (B,S,KP) f16
__global__ void softmax_t_kernel(const float* __restrict__ logits, h16* __restrict__ Hst, int total)
{
    int idx = blockIdx.x * blockDim.x + threadIdx.x;
    if (idx >= total) return;                   // total = B_*S_
    int b = idx / S_;
    int s = idx - b * S_;
    const float* L = logits + ((size_t)b * NCLS) * S_ + s;
    float mx = -3.4e38f, sum = 0.f;
    for (int n = 0; n < NCLS; ++n) {            // single online pass
        float x = L[(size_t)n * S_];
        if (x > mx) { sum *= expf(mx - x); mx = x; }
        sum += expf(x - mx);
    }
    float inv = 1.f / sum;
    h16* o = Hst + (size_t)idx * KP;            // idx*KP*2 = 320B -> 16B aligned
    for (int n0 = 0; n0 < KP; n0 += 8) {
        v8h v;
        #pragma unroll
        for (int u = 0; u < 8; ++u) {
            int n = n0 + u;
            float val = (n < NCLS) ? expf(L[(size_t)n * S_] - mx) * inv : 0.f;
            v[u] = (h16)val;
        }
        *(v8h*)(o + n0) = v;
    }
}

__global__ void concat_kernel(const h16* __restrict__ a, const h16* __restrict__ b,
                              h16* __restrict__ o, int rows)
{
    int idx = blockIdx.x * blockDim.x + threadIdx.x;
    int total = rows * 2 * CD_;
    if (idx >= total) return;
    int r = idx / (2 * CD_);
    int j = idx - r * (2 * CD_);
    o[idx] = (j < CD_) ? a[(size_t)r * CD_ + j] : b[(size_t)r * CD_ + (j - CD_)];
}

__global__ void bias_gelu_kernel(const float* __restrict__ x, const float* __restrict__ bias,
                                 h16* __restrict__ y, int total)
{
    int idx = blockIdx.x * blockDim.x + threadIdx.x;
    if (idx >= total) return;
    int c = idx & (CD_ - 1);
    float v = x[idx] + bias[c];
    float gl = 0.5f * v * (1.f + erff(v * 0.70710678118654752f));   // exact GELU
    y[idx] = (h16)gl;
}

__global__ void combine_kernel(const float* __restrict__ gpre, const float* __restrict__ gb2,
                               const float* __restrict__ cpre, const float* __restrict__ cb2,
                               const float* __restrict__ ce_in, float* __restrict__ ce_out,
                               h16* __restrict__ ce_h, int total)
{
    int idx = blockIdx.x * blockDim.x + threadIdx.x;
    if (idx >= total) return;
    int c = idx & (CD_ - 1);
    float gg = 1.f / (1.f + expf(-(gpre[idx] + gb2[c])));
    float cand = cpre[idx] + cb2[c];
    float ce = gg * cand + (1.f - gg) * ce_in[idx];
    ce_out[idx] = ce;
    ce_h[idx] = (h16)ce;
}

__global__ void make_gbt_kernel(const float* __restrict__ gp, const float* __restrict__ bp,
                                const float* __restrict__ gbias, const float* __restrict__ bbias,
                                h16* __restrict__ gT, h16* __restrict__ bT)
{
    int idx = blockIdx.x * blockDim.x + threadIdx.x;
    const int total = B_ * CF_ * KP;
    if (idx >= total) return;
    int n = idx % KP;
    int c = (idx / KP) % CF_;
    int b = idx / (KP * CF_);
    h16 gv = (h16)0.f, bv = (h16)0.f;
    if (n < NCLS) {
        size_t src = ((size_t)b * NCLS + n) * CF_ + c;
        gv = (h16)(1.f + tanhf(gp[src] + gbias[c]));
        bv = (h16)(bp[src] + bbias[c]);
    }
    gT[idx] = gv;
    bT[idx] = bv;
}

__global__ void film_kernel(const float* __restrict__ F, const float* __restrict__ ge,
                            const float* __restrict__ be, float* __restrict__ out, int total)
{
    int idx = blockIdx.x * blockDim.x + threadIdx.x;
    if (idx >= total) return;
    float f = F[idx];
    out[idx] = (1.f - LAM_) * f + LAM_ * (ge[idx] * f + be[idx]);
}

// ---------------------------------------------------------------------------
extern "C" void kernel_launch(void* const* d_in, const int* in_sizes, int n_in,
                              void* d_out, int out_size, void* d_ws, size_t ws_size,
                              hipStream_t stream)
{
    (void)in_sizes; (void)n_in; (void)out_size; (void)ws_size;
    const float* CE      = (const float*)d_in[0];
    const float* Fmap    = (const float*)d_in[1];
    const float* w_d2f   = (const float*)d_in[2];
    const float* w_f2d   = (const float*)d_in[3];
    const float* gate_w1 = (const float*)d_in[4];
    const float* gate_b1 = (const float*)d_in[5];
    const float* gate_w2 = (const float*)d_in[6];
    const float* gate_b2 = (const float*)d_in[7];
    const float* cand_w1 = (const float*)d_in[8];
    const float* cand_b1 = (const float*)d_in[9];
    const float* cand_w2 = (const float*)d_in[10];
    const float* cand_b2 = (const float*)d_in[11];
    const float* gamma_w = (const float*)d_in[12];
    const float* gamma_b = (const float*)d_in[13];
    const float* beta_w  = (const float*)d_in[14];
    const float* beta_b  = (const float*)d_in[15];
    const float* fln_g   = (const float*)d_in[16];
    const float* fln_b   = (const float*)d_in[17];
    const float* dln_g   = (const float*)d_in[18];
    const float* dln_b   = (const float*)d_in[19];

    float* outCE  = (float*)d_out;                            // (B,N,CD)
    float* outF   = outCE + (size_t)B_ * NCLS * CD_;          // (B,CF,S)
    float* logits = outF + (size_t)B_ * CF_ * S_;             // (B,N,S) == score (TAU=1)

    char* ws = (char*)d_ws;
    size_t off = 0;
    auto alloc = [&](size_t bytes) -> void* {
        void* p = ws + off;
        off += (bytes + 255) & ~(size_t)255;
        return p;
    };

    h16* w_d2f_h = (h16*)alloc((size_t)CF_ * CD_ * 2);
    h16* w_f2d_h = (h16*)alloc((size_t)CD_ * CF_ * 2);
    h16* gw1_h   = (h16*)alloc((size_t)CD_ * 2 * CD_ * 2);
    h16* gw2_h   = (h16*)alloc((size_t)CD_ * CD_ * 2);
    h16* cw1_h   = (h16*)alloc((size_t)CD_ * CD_ * 2);
    h16* cw2_h   = (h16*)alloc((size_t)CD_ * CD_ * 2);
    h16* gmw_h   = (h16*)alloc((size_t)CF_ * CD_ * 2);
    h16* btw_h   = (h16*)alloc((size_t)CF_ * CD_ * 2);

    float* CEn_f    = (float*)alloc((size_t)B_ * NCLS * CD_ * 4);
    h16*   CEn_h    = (h16*)alloc((size_t)B_ * NCLS * CD_ * 2);
    h16*   Fn_t     = (h16*)alloc((size_t)B_ * S_ * CF_ * 2);   // (B,S,CF)
    h16*   CEproj_h = (h16*)alloc((size_t)B_ * NCLS * CF_ * 2);
    h16*   V_h      = (h16*)alloc((size_t)B_ * CD_ * S_ * 2);   // (B,CD,S)
    h16*   Wm       = (h16*)alloc((size_t)B_ * NCLS * S_ * 2);  // (B,N,S)
    float* CnPart   = (float*)alloc((size_t)KSPLIT * B_ * NCLS * CD_ * 4);
    float* Cn_f     = (float*)alloc((size_t)B_ * NCLS * CD_ * 4);
    h16*   Cn_h     = (h16*)alloc((size_t)B_ * NCLS * CD_ * 2);
    h16*   gate_in  = (h16*)alloc((size_t)B_ * NCLS * 2 * CD_ * 2);
    float* h1pre    = (float*)alloc((size_t)B_ * NCLS * CD_ * 4);
    h16*   h1_h     = (h16*)alloc((size_t)B_ * NCLS * CD_ * 2);
    float* gpre     = (float*)alloc((size_t)B_ * NCLS * CD_ * 4);
    float* c1pre    = (float*)alloc((size_t)B_ * NCLS * CD_ * 4);
    h16*   c1_h     = (h16*)alloc((size_t)B_ * NCLS * CD_ * 2);
    float* candpre  = (float*)alloc((size_t)B_ * NCLS * CD_ * 4);
    h16*   CEnew_h  = (h16*)alloc((size_t)B_ * NCLS * CD_ * 2);
    float* gammapre = (float*)alloc((size_t)B_ * NCLS * CF_ * 4);
    float* betapre  = (float*)alloc((size_t)B_ * NCLS * CF_ * 4);
    h16*   gT       = (h16*)alloc((size_t)B_ * CF_ * KP * 2);   // (B,CF,KP)
    h16*   bT       = (h16*)alloc((size_t)B_ * CF_ * KP * 2);
    h16*   Hst      = (h16*)alloc((size_t)B_ * S_ * KP * 2);    // (B,S,KP)
    float* geff     = (float*)alloc((size_t)B_ * CF_ * S_ * 4);
    float* beff     = (float*)alloc((size_t)B_ * CF_ * S_ * 4);

    auto cvt = [&](const float* x, h16* y, int n) {
        cvt_f32_to_f16_kernel<<<(n + 255) / 256, 256, 0, stream>>>(x, y, n);
    };
    cvt(w_d2f,   w_d2f_h, CF_ * CD_);
    cvt(w_f2d,   w_f2d_h, CD_ * CF_);
    cvt(gate_w1, gw1_h,   CD_ * 2 * CD_);
    cvt(gate_w2, gw2_h,   CD_ * CD_);
    cvt(cand_w1, cw1_h,   CD_ * CD_);
    cvt(cand_w2, cw2_h,   CD_ * CD_);
    cvt(gamma_w, gmw_h,   CF_ * CD_);
    cvt(beta_w,  btw_h,   CF_ * CD_);

    ln_rows_kernel<<<B_ * NCLS, 256, 0, stream>>>(CE, dln_g, dln_b, CEn_f, CEn_h, CD_);
    ln_chan_kernel<<<(B_ * S_ + 255) / 256, 256, 0, stream>>>(Fmap, fln_g, fln_b, Fn_t, B_ * S_);

    auto gemm1 = [&](const h16* A, const h16* Bt, float* Cf, h16* Ch,
                     int M, int N, int K, int lda, int ldb, int ldc,
                     long sA, long sB, long sC, float alpha, int mode) {
        dim3 grid((N + 127) / 128, (M + 15) / 16, B_);
        wmma_gemm_nt<<<grid, 256, 0, stream>>>(A, Bt, Cf, Ch, M, N, K,
                                               lda, ldb, ldc, sA, sB, sC, alpha, mode);
    };
    auto gemm2 = [&](const h16* A, const h16* Bt, float* Cf, h16* Ch,
                     int M, int N, int K, int lda, int ldb, int ldc,
                     long sA, long sB, long sC, long sKc, float alpha, int mode, int ksplit) {
        dim3 grid(N / 128, (M + 63) / 64, B_ * ksplit);
        wmma_gemm_nt_2x2<<<grid, 256, 0, stream>>>(A, Bt, Cf, Ch, M, N, K,
                                                   lda, ldb, ldc, sA, sB, sC, sKc,
                                                   alpha, mode, ksplit);
    };

    // CE_proj[b,n,cf] = sum_cd CEn * w_d2f[cf,cd]
    gemm1(CEn_h, w_d2f_h, nullptr, CEproj_h, NCLS, CF_, CD_, CD_, CD_, CF_,
          (long)NCLS * CD_, 0, (long)NCLS * CF_, 1.f, 2);
    // score -> logits: (1/sqrt(CF)) * CE_proj . Fn_t
    gemm2(CEproj_h, Fn_t, logits, nullptr, NCLS, S_, CF_, CF_, CF_, S_,
          (long)NCLS * CF_, (long)S_ * CF_, (long)NCLS * S_, 0, 1.f / 16.f, 1, 1);
    // V[b,d,s] = w_f2d . Fn_t
    gemm2(w_f2d_h, Fn_t, nullptr, V_h, CD_, S_, CF_, CF_, CF_, S_,
          0, (long)S_ * CF_, (long)CD_ * S_, 0, 1.f, 2, 1);

    maskw_kernel<<<B_ * NCLS, 256, 0, stream>>>(logits, Wm);
    softmax_t_kernel<<<(B_ * S_ + 255) / 256, 256, 0, stream>>>(logits, Hst, B_ * S_);

    // Cn[b,n,d] = sum_s Wm * V  (split-K over S for parallelism, then reduce)
    gemm2(Wm, V_h, CnPart, nullptr, NCLS, CD_, S_, S_, S_, CD_,
          (long)NCLS * S_, (long)CD_ * S_, (long)NCLS * CD_,
          (long)B_ * NCLS * CD_, 1.f, 1, KSPLIT);
    reduce_partials_kernel<<<(B_ * NCLS * CD_ + 255) / 256, 256, 0, stream>>>(
        CnPart, Cn_f, Cn_h, B_ * NCLS * CD_, KSPLIT, (long)B_ * NCLS * CD_);

    concat_kernel<<<(B_ * NCLS * 2 * CD_ + 255) / 256, 256, 0, stream>>>(CEn_h, Cn_h, gate_in, B_ * NCLS);

    // gate MLP
    gemm1(gate_in, gw1_h, h1pre, nullptr, NCLS, CD_, 2 * CD_, 2 * CD_, 2 * CD_, CD_,
          (long)NCLS * 2 * CD_, 0, (long)NCLS * CD_, 1.f, 1);
    bias_gelu_kernel<<<(B_ * NCLS * CD_ + 255) / 256, 256, 0, stream>>>(h1pre, gate_b1, h1_h, B_ * NCLS * CD_);
    gemm1(h1_h, gw2_h, gpre, nullptr, NCLS, CD_, CD_, CD_, CD_, CD_,
          (long)NCLS * CD_, 0, (long)NCLS * CD_, 1.f, 1);
    // cand MLP
    gemm1(Cn_h, cw1_h, c1pre, nullptr, NCLS, CD_, CD_, CD_, CD_, CD_,
          (long)NCLS * CD_, 0, (long)NCLS * CD_, 1.f, 1);
    bias_gelu_kernel<<<(B_ * NCLS * CD_ + 255) / 256, 256, 0, stream>>>(c1pre, cand_b1, c1_h, B_ * NCLS * CD_);
    gemm1(c1_h, cw2_h, candpre, nullptr, NCLS, CD_, CD_, CD_, CD_, CD_,
          (long)NCLS * CD_, 0, (long)NCLS * CD_, 1.f, 1);

    combine_kernel<<<(B_ * NCLS * CD_ + 255) / 256, 256, 0, stream>>>(
        gpre, gate_b2, candpre, cand_b2, CE, outCE, CEnew_h, B_ * NCLS * CD_);

    // gamma/beta projections
    gemm1(CEnew_h, gmw_h, gammapre, nullptr, NCLS, CF_, CD_, CD_, CD_, CF_,
          (long)NCLS * CD_, 0, (long)NCLS * CF_, 1.f, 1);
    gemm1(CEnew_h, btw_h, betapre, nullptr, NCLS, CF_, CD_, CD_, CD_, CF_,
          (long)NCLS * CD_, 0, (long)NCLS * CF_, 1.f, 1);
    make_gbt_kernel<<<(B_ * CF_ * KP + 255) / 256, 256, 0, stream>>>(
        gammapre, betapre, gamma_b, beta_b, gT, bT);

    // g_eff/b_eff[b,c,s] = sum_n gT[b,c,n] * Hst[b,s,n]
    gemm2(gT, Hst, geff, nullptr, CF_, S_, KP, KP, KP, S_,
          (long)CF_ * KP, (long)S_ * KP, (long)CF_ * S_, 0, 1.f, 1, 1);
    gemm2(bT, Hst, beff, nullptr, CF_, S_, KP, KP, KP, S_,
          (long)CF_ * KP, (long)S_ * KP, (long)CF_ * S_, 0, 1.f, 1, 1);

    film_kernel<<<(B_ * CF_ * S_ + 255) / 256, 256, 0, stream>>>(Fmap, geff, beff, outF, B_ * CF_ * S_);
}
